// DrugGCN_79955111182532
// MI455X (gfx1250) — matrix operand order repaired and checked by reading, not compiled, not run on hardware
//
#include <hip/hip_runtime.h>

typedef _Float16 half_t;
typedef __attribute__((ext_vector_type(16))) _Float16 v16h;
typedef __attribute__((ext_vector_type(8)))  _Float16 v8h;
typedef __attribute__((ext_vector_type(4)))  _Float16 v4h;
typedef __attribute__((ext_vector_type(8)))  float    v8f;
typedef __attribute__((ext_vector_type(4)))  float    v4f;

#define N_GRAPHS 512
#define NODES    256
#define EPG      4096
#define D        128
#define KINIT    75
#define KPADI    96

union ABf { v16h v; v8h h[2]; };

// ---------------------------------------------------------------------------
// Prep: build f16, column-major (K-contiguous) copies of the weights so a
// WMMA B-fragment per lane is two 16B K-slices, 32B apart.
// ---------------------------------------------------------------------------
__global__ void prep_weights(const float* __restrict__ Wi, const float* __restrict__ W1,
                             const float* __restrict__ W2, half_t* __restrict__ wti,
                             half_t* __restrict__ wt1, half_t* __restrict__ wt2) {
    int tid = blockIdx.x * blockDim.x + threadIdx.x;
    int stride = gridDim.x * blockDim.x;
    for (int i = tid; i < D * D; i += stride) {
        int n = i >> 7, k = i & 127;
        wt1[n * D + k] = (half_t)W1[k * D + n];
        wt2[n * D + k] = (half_t)W2[k * D + n];
    }
    for (int i = tid; i < D * KPADI; i += stride) {
        int n = i / KPADI, k = i - n * KPADI;
        wti[n * KPADI + k] = (k < KINIT) ? (half_t)Wi[k * D + n] : (half_t)0.0f;
    }
}

// ---------------------------------------------------------------------------
// Init transform: h = node_feats @ W_init  (K=75 zero-padded to 96)
// One WG per graph (256 rows), 8 waves, each wave: 2 M-tiles x 8 N-tiles.
// W staged in LDS so B-fragments come from ds_load_b128 (low latency).
// ---------------------------------------------------------------------------
__global__ __launch_bounds__(256) void init_gemm(const float* __restrict__ nf,
                                                 const half_t* __restrict__ wt,
                                                 float* __restrict__ h) {
    __shared__ half_t s_w[D * KPADI];           // 24 KB f16 weight tile
    const int g = blockIdx.x;
    const int t = threadIdx.x;
    const int wave = t >> 5, lane = t & 31;
    const int sel = lane >> 4, lc = lane & 15;
    const long nodebase = (long)g * NODES;

    for (int i = t; i < (D * KPADI) / 4; i += 256)
        ((v4h*)s_w)[i] = ((const v4h*)wt)[i];
    __syncthreads();

    v8f acc[2][8] = {};
    #pragma unroll
    for (int ks = 0; ks < 3; ++ks) {
        const int kb = ks * 32;
        ABf a[2];
        #pragma unroll
        for (int m = 0; m < 2; ++m) {
            const int lrow = (wave * 2 + m) * 16 + lc;
            const float* ar = nf + (nodebase + lrow) * (long)KINIT;
            const int k0 = kb + sel * 8, k1 = kb + 16 + sel * 8;
            v8h lo, hi;
            #pragma unroll
            for (int i = 0; i < 8; ++i) {
                int ka = k0 + i, kx = k1 + i;
                lo[i] = (ka < KINIT) ? (half_t)ar[ka] : (half_t)0.0f;
                hi[i] = (kx < KINIT) ? (half_t)ar[kx] : (half_t)0.0f;
            }
            a[m].h[0] = lo; a[m].h[1] = hi;
        }
        #pragma unroll
        for (int n = 0; n < 8; ++n) {
            const v8h* bp = (const v8h*)(s_w + (n * 16 + lc) * KPADI + kb + sel * 8);
            ABf b; b.h[0] = bp[0]; b.h[1] = bp[2];
            #pragma unroll
            for (int m = 0; m < 2; ++m)
                acc[m][n] = __builtin_amdgcn_wmma_f32_16x16x32_f16(
                    false, a[m].v, false, b.v, (short)0, acc[m][n], false, false);
        }
    }
    #pragma unroll
    for (int m = 0; m < 2; ++m)
        #pragma unroll
        for (int n = 0; n < 8; ++n)
            #pragma unroll
            for (int v = 0; v < 8; ++v) {
                int row = (wave * 2 + m) * 16 + v + 8 * sel;
                h[(nodebase + row) * (long)D + n * 16 + lc] = acc[m][n][v];
            }
}

// ---------------------------------------------------------------------------
// Fused GraphConv layer. LDS plan (130 KB total):
//   s_m   : 64 KB f16 messages (all 256 src rows)
//   s_agg : 64 KB, time-multiplexed: [GEMM phase] first 32 KB = f16 W tile,
//           [scatter phase] f32 agg chunk for half the dst rows
//   s_deg : 2 KB degree counters, reused as f32 norms
// ---------------------------------------------------------------------------
__global__ __launch_bounds__(256) void gcn_layer(const int* __restrict__ esrc,
                                                 const int* __restrict__ edst,
                                                 const half_t* __restrict__ wt,
                                                 const float* __restrict__ bias,
                                                 const float* __restrict__ hin,
                                                 float* __restrict__ hout) {
    __shared__ half_t       s_m[NODES * D];     // 64 KB
    __shared__ float        s_agg[128 * D];     // 64 KB (aliased: W tile during GEMM)
    __shared__ unsigned int s_deg[2 * NODES];   // 2 KB

    const int g = blockIdx.x;
    const int t = threadIdx.x;
    const int wave = t >> 5, lane = t & 31;
    const int sel = lane >> 4, lc = lane & 15;
    const long nodebase = (long)g * NODES;
    const long ebase = (long)g * EPG;
    float*  s_norm = (float*)s_deg;             // [0..255]=norm_out, [256..511]=norm_in
    half_t* s_w    = (half_t*)s_agg;            // 128*128 f16 = 32 KB during GEMM

    // ---- stage W into LDS + zero degree counters
    for (int i = t; i < (D * D) / 4; i += 256)
        ((v4h*)s_w)[i] = ((const v4h*)wt)[i];
    if (t < 2 * NODES) s_deg[t] = 0u;
    __syncthreads();

    // ---- degrees (edges are block-local, so this equals the global segment_sum)
    for (int i = t; i < EPG; i += 256) {
        int s = esrc[ebase + i] & (NODES - 1);
        int d = edst[ebase + i] & (NODES - 1);
        atomicAdd(&s_deg[s], 1u);
        atomicAdd(&s_deg[NODES + d], 1u);
    }
    __syncthreads();
    if (t < 2 * NODES) {
        unsigned dc = s_deg[t];
        s_norm[t] = dc ? rsqrtf((float)dc) : 0.0f;
    }
    __syncthreads();

    // ---- GEMM: s_m = (diag(norm_out) * h) @ W   (f16 WMMA, f32 accum)
    {
        v8f acc[2][8] = {};
        #pragma unroll
        for (int ks = 0; ks < 4; ++ks) {
            const int kb = ks * 32;
            ABf a[2];
            #pragma unroll
            for (int m = 0; m < 2; ++m) {
                const int lrow = (wave * 2 + m) * 16 + lc;
                const float nrm = s_norm[lrow];
                const float* ar = hin + (nodebase + lrow) * (long)D;
                const v4f* p0 = (const v4f*)(ar + kb + sel * 8);
                const v4f* p1 = (const v4f*)(ar + kb + 16 + sel * 8);
                v4f c0 = p0[0], c1 = p0[1], c2 = p1[0], c3 = p1[1];
                v8h lo, hi;
                #pragma unroll
                for (int i = 0; i < 4; ++i) {
                    lo[i]     = (half_t)(c0[i] * nrm);
                    lo[i + 4] = (half_t)(c1[i] * nrm);
                    hi[i]     = (half_t)(c2[i] * nrm);
                    hi[i + 4] = (half_t)(c3[i] * nrm);
                }
                a[m].h[0] = lo; a[m].h[1] = hi;
            }
            #pragma unroll
            for (int n = 0; n < 8; ++n) {
                const v8h* bp = (const v8h*)(s_w + (n * 16 + lc) * D + kb + sel * 8);
                ABf b; b.h[0] = bp[0]; b.h[1] = bp[2];
                #pragma unroll
                for (int m = 0; m < 2; ++m)
                    acc[m][n] = __builtin_amdgcn_wmma_f32_16x16x32_f16(
                        false, a[m].v, false, b.v, (short)0, acc[m][n], false, false);
            }
        }
        #pragma unroll
        for (int m = 0; m < 2; ++m)
            #pragma unroll
            for (int n = 0; n < 8; ++n)
                #pragma unroll
                for (int v = 0; v < 8; ++v) {
                    int row = (wave * 2 + m) * 16 + v + 8 * sel;
                    s_m[row * D + n * 16 + lc] = (half_t)acc[m][n][v];
                }
    }
    __syncthreads();

    // ---- scatter + epilogue, two dst-half passes to bound LDS
    for (int p = 0; p < 2; ++p) {
        for (int i = t; i < 128 * D; i += 256) s_agg[i] = 0.0f;
        __syncthreads();

        const int coff = lane * 4;                    // 4 features per lane
        for (int j = 0; j < EPG / 8; ++j) {           // wave-per-edge
            const long e = ebase + (long)wave * (EPG / 8) + j;
            if ((j & 31) == 0) __builtin_prefetch(&esrc[e + 64], 0, 0);
            const int s = esrc[e] & (NODES - 1);
            const int d = edst[e] & (NODES - 1);
            if ((d >> 7) != p) continue;              // wave-uniform branch
            v4h mv = *(const v4h*)(s_m + s * D + coff);
            float* ap = s_agg + (d & 127) * D + coff;
            atomicAdd(ap + 0, (float)mv[0]);
            atomicAdd(ap + 1, (float)mv[1]);
            atomicAdd(ap + 2, (float)mv[2]);
            atomicAdd(ap + 3, (float)mv[3]);
        }
        __syncthreads();

        const int lr = t >> 1;
        const int row = p * 128 + lr;
        const int cb = (t & 1) * 64;
        const float ni = s_norm[NODES + row];
        float* orow = hout + (nodebase + row) * (long)D;
        #pragma unroll
        for (int c = cb; c < cb + 64; c += 4) {
            v4f v;
            #pragma unroll
            for (int q = 0; q < 4; ++q) {
                float val = s_agg[lr * D + c + q] * ni + bias[c + q];
                v[q] = val > 0.0f ? val : 0.0f;
            }
            *(v4f*)(orow + c) = v;
        }
        __syncthreads();
    }
}

// ---------------------------------------------------------------------------
extern "C" void kernel_launch(void* const* d_in, const int* in_sizes, int n_in,
                              void* d_out, int out_size, void* d_ws, size_t ws_size,
                              hipStream_t stream) {
    (void)in_sizes; (void)n_in; (void)out_size; (void)ws_size;
    const float* node_feats = (const float*)d_in[0];
    const int*   esrc       = (const int*)d_in[1];
    const int*   edst       = (const int*)d_in[2];
    const float* Wi         = (const float*)d_in[3];
    const float* W1         = (const float*)d_in[4];
    const float* b1         = (const float*)d_in[5];
    const float* W2         = (const float*)d_in[6];
    const float* b2         = (const float*)d_in[7];
    float* out = (float*)d_out;

    char* ws = (char*)d_ws;
    half_t* wt1 = (half_t*)(ws);            // 128*128 f16 = 32 KB
    half_t* wt2 = (half_t*)(ws + 32768);    // 32 KB
    half_t* wti = (half_t*)(ws + 65536);    // 128*96 f16 = 24 KB
    float*  h   = (float*)(ws + 131072);    // 131072*128 f32 = 64 MB

    prep_weights<<<64, 256, 0, stream>>>(Wi, W1, W2, wti, wt1, wt2);
    init_gemm<<<N_GRAPHS, 256, 0, stream>>>(node_feats, wti, h);
    gcn_layer<<<N_GRAPHS, 256, 0, stream>>>(esrc, edst, wt1, b1, h, h);    // in-place
    gcn_layer<<<N_GRAPHS, 256, 0, stream>>>(esrc, edst, wt2, b2, h, out);
}
